// ThinkingRNN_21440476741882
// MI455X (gfx1250) — compile-verified, matrix operand order checked by reading, not run on hardware
//
#include <hip/hip_runtime.h>
#include <math.h>

typedef __bf16 bf16;
typedef __attribute__((ext_vector_type(16))) __bf16 bf16x16;
typedef __attribute__((ext_vector_type(8)))  __bf16 bf16x8;
typedef __attribute__((ext_vector_type(8)))  float  f32x8;

#define B_    1024
#define T_    67
#define H_    512
#define V_    128
#define N_    64
#define TTH   17
#define THINK_END 127
#define MAX_THINK 16
#define EPSF  1e-10f

enum { MODE_ENC = 0, MODE_THINK = 1, MODE_GEN = 2 };

// ---------------- one-time conversion / init ----------------
__global__ void k_f32_to_bf16(const float* __restrict__ src, bf16* __restrict__ dst, int n) {
  int i = blockIdx.x * blockDim.x + threadIdx.x;
  if (i < n) dst[i] = (bf16)src[i];
}

__global__ void k_init(bf16* __restrict__ h, int* __restrict__ halted, int* __restrict__ steps) {
  int i = blockIdx.x * blockDim.x + threadIdx.x;
  if (i < B_ * H_) h[i] = (bf16)0.0f;
  if (i < B_) { halted[i] = 0; steps[i] = 0; }
}

__global__ void k_build_cur(const float* __restrict__ x, const int* __restrict__ lengths,
                            bf16* __restrict__ cur) {
  int i = blockIdx.x * blockDim.x + threadIdx.x;
  if (i >= B_ * V_) return;
  int b = i / V_, v = i % V_;
  int t = lengths[b] - 1;
  cur[i] = (bf16)x[((size_t)b * T_ + t) * V_ + v];
}

// ---------------- WMMA GEMM core: 2x2 register-blocked ----------------
// Per-lane fragment layout per CDNA5 ISA 16-bit A/B: lane half selects K {0..7,16..23}
// vs {8..15,24..31}; each half is two contiguous 16B runs -> global_load_b128.
// 4 WMMAs per K-chunk from 2 A-frags + 2 B-frags (both operands reused in-register).
__device__ __forceinline__ bf16x16 load_frag(const bf16* p) {
  bf16x8 lo = *(const bf16x8*)(p);
  bf16x8 hi = *(const bf16x8*)(p + 16);
  return __builtin_shufflevector(lo, hi, 0,1,2,3,4,5,6,7,8,9,10,11,12,13,14,15);
}

template <int K>
__device__ __forceinline__ void gemm_accum4(const bf16* __restrict__ A, int lda,
                                            const bf16* __restrict__ W, int ldw,
                                            int r0, int r1, int c0, int c1, int kh,
                                            f32x8& acc00, f32x8& acc01,
                                            f32x8& acc10, f32x8& acc11) {
  const bf16* pa0 = A + (size_t)r0 * lda + kh;
  const bf16* pa1 = A + (size_t)r1 * lda + kh;
  const bf16* pb0 = W + (size_t)c0 * ldw + kh;
  const bf16* pb1 = W + (size_t)c1 * ldw + kh;
#pragma unroll 2
  for (int k0 = 0; k0 < K; k0 += 32) {
    bf16x16 a0 = load_frag(pa0 + k0);
    bf16x16 a1 = load_frag(pa1 + k0);
    bf16x16 b0 = load_frag(pb0 + k0);
    bf16x16 b1 = load_frag(pb1 + k0);
    acc00 = __builtin_amdgcn_wmma_f32_16x16x32_bf16(false, a0, false, b0, (short)0, acc00, false, false);
    acc01 = __builtin_amdgcn_wmma_f32_16x16x32_bf16(false, a0, false, b1, (short)0, acc01, false, false);
    acc10 = __builtin_amdgcn_wmma_f32_16x16x32_bf16(false, a1, false, b0, (short)0, acc10, false, false);
    acc11 = __builtin_amdgcn_wmma_f32_16x16x32_bf16(false, a1, false, b1, (short)0, acc11, false, false);
  }
}

// ---------------- RNN cell step: hn = tanh(A1@W1^T + h@W2^T + b_ih + b_hh) ----------------
// grid (B/128, H/64), block 256 (8 waves). Wave -> 32 rows x 32 cols (2x2 WMMA tiles).
__global__ __launch_bounds__(256) void k_rnn_step(
    const bf16* __restrict__ A1, int lda1,            // B x V activations (x_t or cur)
    const bf16* __restrict__ w1,                      // H x V  (row-major => B-operand ready)
    const bf16* __restrict__ h_in,                    // B x H
    const bf16* __restrict__ w2,                      // H x H
    const float* __restrict__ b_ih, const float* __restrict__ b_hh,
    bf16* __restrict__ hn_out,                        // raw hn (THINK mode)
    bf16* __restrict__ h_out,                         // predicated h
    int mode, int t,
    const int* __restrict__ lengths, const int* __restrict__ halted) {
  int lane = threadIdx.x & 31;
  int wave = threadIdx.x >> 5;
  int mb = blockIdx.x * 128 + (wave & 3) * 32;   // two 16-row tiles: mb, mb+16
  int nb = blockIdx.y * 64 + (wave >> 2) * 32;   // two 16-col tiles: nb, nb+16
  int kh = (lane >> 4) * 8;
  int r0 = mb + (lane & 15);
  int r1 = r0 + 16;
  int c0 = nb + (lane & 15);
  int c1 = c0 + 16;

  f32x8 a00 = {}, a01 = {}, a10 = {}, a11 = {};
  gemm_accum4<V_>(A1, lda1, w1, V_, r0, r1, c0, c1, kh, a00, a01, a10, a11);
  gemm_accum4<H_>(h_in, H_, w2, H_, r0, r1, c0, c1, kh, a00, a01, a10, a11);

  float bias0 = b_ih[c0] + b_hh[c0];
  float bias1 = b_ih[c1] + b_hh[c1];
  int rb0 = mb + ((lane >> 4) << 3);       // rows for acc row-tile 0
  int rb1 = rb0 + 16;                      // rows for acc row-tile 1
#pragma unroll
  for (int r = 0; r < 8; ++r) {
    int rowA = rb0 + r;
    int rowB = rb1 + r;
    float v00 = tanhf(a00[r] + bias0);
    float v01 = tanhf(a01[r] + bias1);
    float v10 = tanhf(a10[r] + bias0);
    float v11 = tanhf(a11[r] + bias1);
    bf16 o00 = (bf16)v00, o01 = (bf16)v01, o10 = (bf16)v10, o11 = (bf16)v11;
    int iA0 = rowA * H_ + c0, iA1 = rowA * H_ + c1;
    int iB0 = rowB * H_ + c0, iB1 = rowB * H_ + c1;
    if (mode == MODE_GEN) {
      h_out[iA0] = o00; h_out[iA1] = o01;
      h_out[iB0] = o10; h_out[iB1] = o11;
    } else if (mode == MODE_ENC) {
      bool kA = t < lengths[rowA];
      bool kB = t < lengths[rowB];
      h_out[iA0] = kA ? o00 : h_in[iA0];
      h_out[iA1] = kA ? o01 : h_in[iA1];
      h_out[iB0] = kB ? o10 : h_in[iB0];
      h_out[iB1] = kB ? o11 : h_in[iB1];
    } else { // THINK
      hn_out[iA0] = o00; hn_out[iA1] = o01;
      hn_out[iB0] = o10; hn_out[iB1] = o11;
      bool aA = (halted[rowA] == 0);
      bool aB = (halted[rowB] == 0);
      h_out[iA0] = aA ? o00 : h_in[iA0];
      h_out[iA1] = aA ? o01 : h_in[iA1];
      h_out[iB0] = aB ? o10 : h_in[iB0];
      h_out[iB1] = aB ? o11 : h_in[iB1];
    }
  }
}

// ---------------- logits = hn @ w_fc^T + b_fc ---------------- grid (B/128, V/64)
__global__ __launch_bounds__(256) void k_logits(
    const bf16* __restrict__ hn, const bf16* __restrict__ w_fc,
    const float* __restrict__ b_fc, float* __restrict__ logits) {
  int lane = threadIdx.x & 31;
  int wave = threadIdx.x >> 5;
  int mb = blockIdx.x * 128 + (wave & 3) * 32;
  int nb = blockIdx.y * 64 + (wave >> 2) * 32;
  int kh = (lane >> 4) * 8;
  int r0 = mb + (lane & 15);
  int r1 = r0 + 16;
  int c0 = nb + (lane & 15);
  int c1 = c0 + 16;
  f32x8 a00 = {}, a01 = {}, a10 = {}, a11 = {};
  gemm_accum4<H_>(hn, H_, w_fc, H_, r0, r1, c0, c1, kh, a00, a01, a10, a11);
  float bias0 = b_fc[c0], bias1 = b_fc[c1];
  int rb0 = mb + ((lane >> 4) << 3);
  int rb1 = rb0 + 16;
#pragma unroll
  for (int r = 0; r < 8; ++r) {
    int rowA = rb0 + r;
    int rowB = rb1 + r;
    logits[rowA * V_ + c0] = a00[r] + bias0;
    logits[rowA * V_ + c1] = a01[r] + bias1;
    logits[rowB * V_ + c0] = a10[r] + bias0;
    logits[rowB * V_ + c1] = a11[r] + bias1;
  }
}

// ---------------- Gumbel straight-through sampling ----------------
__device__ __forceinline__ float wave_max32(float v) {
#pragma unroll
  for (int m = 16; m; m >>= 1) v = fmaxf(v, __shfl_xor(v, m, 32));
  return v;
}
__device__ __forceinline__ float wave_sum32(float v) {
#pragma unroll
  for (int m = 16; m; m >>= 1) v += __shfl_xor(v, m, 32);
  return v;
}

// One wave per batch row; lane covers 4 vocab entries. grid(B/8), block 256.
__device__ __forceinline__ void gumbel_row(const float* __restrict__ lr,
                                           const float* __restrict__ ur,
                                           int lane, float s[4], float e[4],
                                           float& sum, int& tok) {
  int v0 = lane * 4;
#pragma unroll
  for (int j = 0; j < 4; ++j) {
    float uu = ur[v0 + j];
    float g = -logf(-logf(uu + EPSF) + EPSF);
    s[j] = lr[v0 + j] + g;
  }
  float m = fmaxf(fmaxf(s[0], s[1]), fmaxf(s[2], s[3]));
  m = wave_max32(m);
  sum = 0.0f;
#pragma unroll
  for (int j = 0; j < 4; ++j) { e[j] = expf(s[j] - m); sum += e[j]; }
  sum = wave_sum32(sum);
  // argmax with first-index tie-break (jnp.argmax semantics)
  float bv = s[0]; int bi = v0;
#pragma unroll
  for (int j = 1; j < 4; ++j) if (s[j] > bv) { bv = s[j]; bi = v0 + j; }
#pragma unroll
  for (int m2 = 16; m2; m2 >>= 1) {
    float ov = __shfl_xor(bv, m2, 32);
    int   oi = __shfl_xor(bi, m2, 32);
    if (ov > bv || (ov == bv && oi < bi)) { bv = ov; bi = oi; }
  }
  tok = bi;
}

__global__ __launch_bounds__(256) void k_sample_think(
    const float* __restrict__ logits, const float* __restrict__ u,
    bf16* __restrict__ cur, int* __restrict__ halted, int* __restrict__ steps, int kstep) {
  int lane = threadIdx.x & 31;
  int b = blockIdx.x * 8 + (threadIdx.x >> 5);
  float s[4], e[4], sum; int tok;
  gumbel_row(logits + b * V_, u + b * V_, lane, s, e, sum, tok);
  bool active   = (halted[b] == 0);
  bool halt_now = (tok == THINK_END) || (kstep + 1 > MAX_THINK);
  int v0 = lane * 4;
#pragma unroll
  for (int j = 0; j < 4; ++j) {
    float ys = e[j] / sum;
    float yh = (v0 + j == tok) ? 1.0f : 0.0f;
    float y  = ys + (yh - ys);
    if (active) cur[b * V_ + v0 + j] = (bf16)y;
  }
  if (lane == 0 && active) {
    steps[b] += 1;
    if (halt_now) halted[b] = 1;
  }
}

__global__ __launch_bounds__(256) void k_sample_gen(
    const float* __restrict__ logits, const float* __restrict__ u,
    bf16* __restrict__ cur, float* __restrict__ out, int nstep) {
  int lane = threadIdx.x & 31;
  int b = blockIdx.x * 8 + (threadIdx.x >> 5);
  float s[4], e[4], sum; int tok;
  gumbel_row(logits + b * V_, u + b * V_, lane, s, e, sum, tok);
  int v0 = lane * 4;
  float* orow = out + (size_t)b * N_ * (V_ - 1) + (size_t)nstep * (V_ - 1);
#pragma unroll
  for (int j = 0; j < 4; ++j) {
    int v = v0 + j;
    float ys = e[j] / sum;
    float yh = (v == tok) ? 1.0f : 0.0f;
    float y  = ys + (yh - ys);
    cur[b * V_ + v] = (bf16)y;
    if (v < V_ - 1) orow[v] = y;
  }
}

__global__ void k_write_steps(const int* __restrict__ steps, float* __restrict__ out) {
  int i = blockIdx.x * blockDim.x + threadIdx.x;
  if (i < B_) out[i] = (float)steps[i];
}

// ---------------- host orchestration ----------------
extern "C" void kernel_launch(void* const* d_in, const int* in_sizes, int n_in,
                              void* d_out, int out_size, void* d_ws, size_t ws_size,
                              hipStream_t stream) {
  (void)in_sizes; (void)n_in; (void)out_size; (void)ws_size;
  const float* x      = (const float*)d_in[0];
  const int*   lengths= (const int*)d_in[1];
  const float* w_ih   = (const float*)d_in[2];
  const float* w_hh   = (const float*)d_in[3];
  const float* b_ih   = (const float*)d_in[4];
  const float* b_hh   = (const float*)d_in[5];
  const float* w_fc   = (const float*)d_in[6];
  const float* b_fc   = (const float*)d_in[7];
  const float* u_think= (const float*)d_in[8];
  const float* u_gen  = (const float*)d_in[9];
  float* out = (float*)d_out;

  char* ws = (char*)d_ws;
  size_t off = 0;
  auto walloc = [&](size_t bytes) -> void* {
    void* p = ws + off;
    off = (off + bytes + 255) & ~(size_t)255;
    return p;
  };
  bf16* xb     = (bf16*)walloc((size_t)B_ * T_ * V_ * 2);
  bf16* wihb   = (bf16*)walloc((size_t)H_ * V_ * 2);
  bf16* whhb   = (bf16*)walloc((size_t)H_ * H_ * 2);
  bf16* wfcb   = (bf16*)walloc((size_t)V_ * H_ * 2);
  bf16* h0     = (bf16*)walloc((size_t)B_ * H_ * 2);
  bf16* h1     = (bf16*)walloc((size_t)B_ * H_ * 2);
  bf16* hnbuf  = (bf16*)walloc((size_t)B_ * H_ * 2);
  bf16* cur    = (bf16*)walloc((size_t)B_ * V_ * 2);
  float* logits= (float*)walloc((size_t)B_ * V_ * 4);
  int* halted  = (int*)walloc((size_t)B_ * 4);
  int* steps   = (int*)walloc((size_t)B_ * 4);

  // one-time conversions + init
  k_f32_to_bf16<<<(B_*T_*V_ + 255)/256, 256, 0, stream>>>(x, xb, B_*T_*V_);
  k_f32_to_bf16<<<(H_*V_ + 255)/256, 256, 0, stream>>>(w_ih, wihb, H_*V_);
  k_f32_to_bf16<<<(H_*H_ + 255)/256, 256, 0, stream>>>(w_hh, whhb, H_*H_);
  k_f32_to_bf16<<<(V_*H_ + 255)/256, 256, 0, stream>>>(w_fc, wfcb, V_*H_);
  k_init<<<(B_*H_ + 255)/256, 256, 0, stream>>>(h0, halted, steps);

  dim3 blk(256);
  dim3 gH(B_/128, H_/64);   // (8, 8)
  dim3 gV(B_/128, V_/64);   // (8, 2)

  bf16* hp = h0; bf16* hq = h1;

  // encoder scan
  for (int t = 0; t < T_; ++t) {
    k_rnn_step<<<gH, blk, 0, stream>>>(xb + (size_t)t * V_, T_ * V_, wihb, hp, whhb,
                                       b_ih, b_hh, hq, hq, MODE_ENC, t, lengths, halted);
    bf16* tmp = hp; hp = hq; hq = tmp;
  }

  k_build_cur<<<(B_*V_ + 255)/256, 256, 0, stream>>>(x, lengths, cur);

  // think scan
  for (int k = 0; k < TTH; ++k) {
    k_rnn_step<<<gH, blk, 0, stream>>>(cur, V_, wihb, hp, whhb, b_ih, b_hh,
                                       hnbuf, hq, MODE_THINK, 0, lengths, halted);
    k_logits<<<gV, blk, 0, stream>>>(hnbuf, wfcb, b_fc, logits);
    k_sample_think<<<B_/8, blk, 0, stream>>>(logits, u_think + (size_t)k * B_ * V_,
                                             cur, halted, steps, k);
    bf16* tmp = hp; hp = hq; hq = tmp;
  }

  // gen scan
  for (int n = 0; n < N_; ++n) {
    k_rnn_step<<<gH, blk, 0, stream>>>(cur, V_, wihb, hp, whhb, b_ih, b_hh,
                                       hq, hq, MODE_GEN, 0, lengths, halted);
    k_logits<<<gV, blk, 0, stream>>>(hq, wfcb, b_fc, logits);
    k_sample_gen<<<B_/8, blk, 0, stream>>>(logits, u_gen + (size_t)n * B_ * V_,
                                           cur, out, n);
    bf16* tmp = hp; hp = hq; hq = tmp;
  }

  k_write_steps<<<(B_ + 255)/256, 256, 0, stream>>>(steps, out + (size_t)B_ * N_ * (V_ - 1));
}
